// Dual_CrossAttention_Conv_87840671138517
// MI455X (gfx1250) — matrix operand order repaired and checked by reading, not compiled, not run on hardware
//
#include <hip/hip_runtime.h>
#include <hip/hip_bf16.h>

typedef __bf16 bf16;
typedef __attribute__((ext_vector_type(16))) __bf16 v16bf;
typedef __attribute__((ext_vector_type(8)))  float  v8f;

#define WMMA_BF16(a, b, c) \
  __builtin_amdgcn_wmma_f32_16x16x32_bf16(false, (a), false, (b), (short)0, (c), false, false)

__device__ __forceinline__ v8f v8f_zero() {
  v8f z = {0.f, 0.f, 0.f, 0.f, 0.f, 0.f, 0.f, 0.f};
  return z;
}

// Fragment from 32 contiguous bf16-bytes (two adjacent b128 loads).
__device__ __forceinline__ v16bf load_frag_c32(const bf16* p) {
  union { uint4 q[2]; v16bf v; } u;
  const uint4* s = (const uint4*)p;
  u.q[0] = s[0];
  u.q[1] = s[1];
  return u.v;
}

// Fragment from two separate 16-byte runs (A-fragment: k runs hf*16 and hf*16+8).
__device__ __forceinline__ v16bf load_frag_2x16(const bf16* p0, const bf16* p1) {
  union { uint4 q[2]; v16bf v; } u;
  u.q[0] = *(const uint4*)p0;
  u.q[1] = *(const uint4*)p1;
  return u.v;
}

__device__ __forceinline__ unsigned pack2_bf16(float a, float b) {
  union { bf16 h[2]; unsigned u; } x;
  x.h[0] = (bf16)a;
  x.h[1] = (bf16)b;
  return x.u;
}

// A fragment 16x32 (MxK) from K-major source (A[m][k] at src[k*strideK + m]).
// Used once per workgroup for Q (global, strided) - amortized over 128 iters.
__device__ __forceinline__ v16bf load_A_kmajor(const bf16* src, int strideK, int lane) {
  const int hf = lane >> 4, r = lane & 15;
  v16bf a;
#pragma unroll
  for (int e = 0; e < 16; ++e) {
    const int kl = e + (hf << 4);
    a[e] = src[(size_t)kl * strideK + r];
  }
  return a;
}

// Weight A-fragment from global f32 row-major [M][K]:
// two contiguous 8-float runs per lane -> 4x float4 + pack to bf16.
__device__ __forceinline__ v16bf load_A_w_f32(const float* base, int strideM, int lane) {
  const int hf = lane >> 4, r = lane & 15;
  const float* p = base + (size_t)r * strideM + (hf << 4);
  const float4 a0 = ((const float4*)p)[0];
  const float4 a1 = ((const float4*)p)[1];
  const float4 b0 = ((const float4*)(p + 8))[0];
  const float4 b1 = ((const float4*)(p + 8))[1];
  v16bf v;
  v[0] = (bf16)a0.x; v[1] = (bf16)a0.y; v[2] = (bf16)a0.z; v[3] = (bf16)a0.w;
  v[4] = (bf16)a1.x; v[5] = (bf16)a1.y; v[6] = (bf16)a1.z; v[7] = (bf16)a1.w;
  v[8]  = (bf16)b0.x; v[9]  = (bf16)b0.y; v[10] = (bf16)b0.z; v[11] = (bf16)b0.w;
  v[12] = (bf16)b1.x; v[13] = (bf16)b1.y; v[14] = (bf16)b1.z; v[15] = (bf16)b1.w;
  return v;
}

// ============================================================================
// conv1x1:  Y[b,o,n] = sum_c W[o,c]*X[b,c,n] + bias[o]   (+ optional residual)
// 128 threads (4 waves), 64x64 tile, K-step 32. Weight A-fragments preloaded
// into registers; X staged N-major in LDS (contiguous 32B B-fragments).
// Staging is register-buffered: all global loads issue before any LDS drain.
// Nsp is compile-time (4096) so addressing is shift/offset only.
// ============================================================================
template <int CIN, bool XBF16, bool RES>
__global__ __launch_bounds__(128) void conv1x1_kernel(
    const float* __restrict__ Wm, const float* __restrict__ bias,
    const void* __restrict__ Xv, const float* __restrict__ Xres,
    void* __restrict__ Yv, int Cout) {
  constexpr int Nsp = 4096;  // H*W, fixed for this problem
  constexpr int XP = 40;     // padded pitch (k) of N-major X tile
  const int tid = threadIdx.x, wave = tid >> 5, lane = tid & 31;
  const int hf = lane >> 4, c16 = lane & 15;
  const int n0 = blockIdx.x * 64;
  const int m0 = blockIdx.y * 64;
  const int bb = blockIdx.z;

  __shared__ __align__(16) bf16 Xts[64 * XP];  // [n][k]

  // preload all weight fragments for this wave's 16 output rows
  v16bf wfrag[CIN / 32];
#pragma unroll
  for (int kc = 0; kc < CIN / 32; ++kc)
    wfrag[kc] = load_A_w_f32(Wm + (size_t)(m0 + wave * 16) * CIN + kc * 32, CIN, lane);

  v8f acc[4];
#pragma unroll
  for (int i = 0; i < 4; ++i) acc[i] = v8f_zero();

#pragma unroll
  for (int kc = 0; kc < CIN / 32; ++kc) {
    __syncthreads();
    // stage 32(k) x 64(n) tile transposed to [n][k] bf16:
    // phase 1 - issue all global loads into registers; phase 2 - drain to LDS.
    if (XBF16) {
      const bf16* Xb = (const bf16*)Xv + (size_t)bb * CIN * Nsp;
      uint4 xr[2];
#pragma unroll
      for (int it = 0; it < 2; ++it) {
        const int i = tid + it * 128;
        const int k = i >> 3, j = i & 7;
        xr[it] = ((const uint4*)(Xb + (size_t)(kc * 32 + k) * Nsp + n0))[j];
      }
#pragma unroll
      for (int it = 0; it < 2; ++it) {
        const int i = tid + it * 128;
        const int k = i >> 3, j = i & 7;
        union { uint4 q; bf16 h[8]; } d;
        d.q = xr[it];
#pragma unroll
        for (int t = 0; t < 8; ++t) Xts[(8 * j + t) * XP + k] = d.h[t];
      }
    } else {
      const float* Xb = (const float*)Xv + (size_t)bb * CIN * Nsp;
      float4 xr[4];
#pragma unroll
      for (int it = 0; it < 4; ++it) {
        const int i = tid + it * 128;
        const int k = i >> 4, j = i & 15;
        xr[it] = ((const float4*)(Xb + (size_t)(kc * 32 + k) * Nsp + n0))[j];
      }
#pragma unroll
      for (int it = 0; it < 4; ++it) {
        const int i = tid + it * 128;
        const int k = i >> 4, j = i & 15;
        Xts[(4 * j + 0) * XP + k] = (bf16)xr[it].x;
        Xts[(4 * j + 1) * XP + k] = (bf16)xr[it].y;
        Xts[(4 * j + 2) * XP + k] = (bf16)xr[it].z;
        Xts[(4 * j + 3) * XP + k] = (bf16)xr[it].w;
      }
    }
    __syncthreads();
#pragma unroll
    for (int nf = 0; nf < 4; ++nf) {
      const v16bf b = load_frag_c32(Xts + (nf * 16 + c16) * XP + (hf << 4));
      acc[nf] = WMMA_BF16(wfrag[kc], b, acc[nf]);
    }
  }

  float bv[8];
#pragma unroll
  for (int r = 0; r < 8; ++r) bv[r] = bias[m0 + wave * 16 + r + (hf << 3)];

  if (RES) {
    float* Y = (float*)Yv + (size_t)bb * Cout * Nsp;
    const float* Xr = Xres + (size_t)bb * Cout * Nsp;
#pragma unroll
    for (int nf = 0; nf < 4; ++nf)
#pragma unroll
      for (int r = 0; r < 8; ++r) {
        const int m = m0 + wave * 16 + r + (hf << 3);
        const int n = n0 + nf * 16 + c16;
        const size_t idx = (size_t)m * Nsp + n;
        Y[idx] = Xr[idx] + acc[nf][r] + bv[r];
      }
  } else {
    bf16* Y = (bf16*)Yv + (size_t)bb * Cout * Nsp;
#pragma unroll
    for (int nf = 0; nf < 4; ++nf)
#pragma unroll
      for (int r = 0; r < 8; ++r) {
        const int m = m0 + wave * 16 + r + (hf << 3);
        const int n = n0 + nf * 16 + c16;
        Y[(size_t)m * Nsp + n] = (bf16)(acc[nf][r] + bv[r]);
      }
  }
}

// ============================================================================
// Flash attention over [C][N] channel-major bf16 Q,K,V; online softmax over
// N=4096 keys; O bf16 [C][N]. 128 threads = 4 waves, 16 query rows per wave.
// K staged transposed [key][chan]; V native [chan][key] is already N-major.
// K/V staging software-pipelined through registers with double-buffered LDS.
// ============================================================================
template <int C>
__global__ __launch_bounds__(128) void flash_attn_kernel(
    const bf16* __restrict__ Qg, const bf16* __restrict__ Kg,
    const bf16* __restrict__ Vg, bf16* __restrict__ Og) {
  constexpr int N = 4096, Br = 64, Bc = 32, NW = 4;
  constexpr int NB = N / Bc;   // 128 key blocks
  constexpr int CP = C + 8;    // pitch of Kts rows (chan)
  constexpr int VP = Bc + 8;   // pitch of Vs rows (key)
  constexpr int PP = 40;       // pitch of P rows (k)
  constexpr int SSP = 36;      // pitch of S rows (f32)
  constexpr int NSTG = C / 32; // staging uint4-chunks per thread
  const int tid = threadIdx.x, wave = tid >> 5, lane = tid & 31;
  const int hf = lane >> 4, c16 = lane & 15;
  const int bb = blockIdx.y;
  const int n0 = blockIdx.x * Br;
  const int mW = wave * 16;

  const bf16* Qb = Qg + (size_t)bb * C * N;
  const bf16* Kb = Kg + (size_t)bb * C * N;
  const bf16* Vb = Vg + (size_t)bb * C * N;
  bf16* Ob = Og + (size_t)bb * C * N;

  __shared__ __align__(16) bf16 Kts[2 * Bc * CP];   // double-buffered [key][chan]
  __shared__ __align__(16) bf16 Vs[2 * C * VP];     // double-buffered [chan][key]
  __shared__ __align__(16) float Ss[NW * 16 * SSP]; // per-wave scores
  __shared__ __align__(16) bf16 Ps[NW * 16 * PP];   // per-wave probs [m][k]
  __shared__ float rowM[Br], rowL[Br], rowA[Br];

  if (tid < Br) { rowM[tid] = -3.0e38f; rowL[tid] = 0.f; }

  // Q A-fragments: invariant across key blocks, keep in registers.
  v16bf qf[C / 32];
#pragma unroll
  for (int kc = 0; kc < C / 32; ++kc)
    qf[kc] = load_A_kmajor(Qb + (size_t)kc * 32 * N + n0 + mW, N, lane);

  v8f oacc[C / 16];
#pragma unroll
  for (int i = 0; i < C / 16; ++i) oacc[i] = v8f_zero();

  const float scale = 1.0f / sqrtf((float)C);
  float* Sw = Ss + wave * 16 * SSP;
  bf16* Pw = Ps + wave * 16 * PP;

  uint4 kreg[NSTG], vreg[NSTG];
  // prologue: stage block 0 into buffer 0
#pragma unroll
  for (int it = 0; it < NSTG; ++it) {
    const int i = tid + it * 128;
    const int c = i >> 2, j = i & 3;
    kreg[it] = ((const uint4*)(Kb + (size_t)c * N))[j];
    vreg[it] = ((const uint4*)(Vb + (size_t)c * N))[j];
  }
#pragma unroll
  for (int it = 0; it < NSTG; ++it) {
    const int i = tid + it * 128;
    const int c = i >> 2, j = i & 3;
    ((uint4*)(Vs + c * VP))[j] = vreg[it];
    union { uint4 q; bf16 h[8]; } kd;
    kd.q = kreg[it];
#pragma unroll
    for (int t = 0; t < 8; ++t) Kts[(8 * j + t) * CP + c] = kd.h[t];
  }
  __syncthreads();

  for (int kb = 0; kb < NB; ++kb) {
    const int cur = kb & 1;
    const bf16* Kc = Kts + cur * (Bc * CP);
    const bf16* Vc = Vs + cur * (C * VP);

    // issue next block's global loads (latency hidden under compute below)
    if (kb + 1 < NB) {
      const int nk = (kb + 1) * Bc;
#pragma unroll
      for (int it = 0; it < NSTG; ++it) {
        const int i = tid + it * 128;
        const int c = i >> 2, j = i & 3;
        kreg[it] = ((const uint4*)(Kb + (size_t)c * N + nk))[j];
        vreg[it] = ((const uint4*)(Vb + (size_t)c * N + nk))[j];
      }
    }

    // ---- S = Q^T K : 16 x 32 per wave ----
    v8f sf0 = v8f_zero(), sf1 = v8f_zero();
#pragma unroll
    for (int kc = 0; kc < C / 32; ++kc) {
      const v16bf b0 = load_frag_c32(Kc + (0 + c16) * CP + kc * 32 + (hf << 4));
      const v16bf b1 = load_frag_c32(Kc + (16 + c16) * CP + kc * 32 + (hf << 4));
      sf0 = WMMA_BF16(qf[kc], b0, sf0);
      sf1 = WMMA_BF16(qf[kc], b1, sf1);
    }
#pragma unroll
    for (int r = 0; r < 8; ++r) {
      const int m = r + (hf << 3);
      Sw[m * SSP + c16] = sf0[r] * scale;
      Sw[m * SSP + 16 + c16] = sf1[r] * scale;
    }
    __syncthreads();

    // ---- online softmax: lanes 0..15 each own one query row ----
    if (lane < 16) {
      const int row = mW + lane;
      const float4* S4 = (const float4*)(Ss + (wave * 16 + lane) * SSP);
      float4 sv[8];
#pragma unroll
      for (int j = 0; j < 8; ++j) sv[j] = S4[j];
      const float mold = rowM[row];
      float mx = mold;
#pragma unroll
      for (int j = 0; j < 8; ++j)
        mx = fmaxf(mx, fmaxf(fmaxf(sv[j].x, sv[j].y), fmaxf(sv[j].z, sv[j].w)));
      const float alpha = __expf(mold - mx);
      float s = 0.f;
      unsigned* Pr = (unsigned*)(Pw + lane * PP);
#pragma unroll
      for (int j = 0; j < 8; ++j) {
        const float p0 = __expf(sv[j].x - mx);
        const float p1 = __expf(sv[j].y - mx);
        const float p2 = __expf(sv[j].z - mx);
        const float p3 = __expf(sv[j].w - mx);
        s += (p0 + p1) + (p2 + p3);
        Pr[2 * j + 0] = pack2_bf16(p0, p1);
        Pr[2 * j + 1] = pack2_bf16(p2, p3);
      }
      rowL[row] = rowL[row] * alpha + s;
      rowM[row] = mx;
      rowA[row] = alpha;
    }
    __syncthreads();

    // ---- O = diag(alpha) * O + P @ V^T ----
    const bf16* Pa = Pw + ((lane & 15) * PP) + (hf << 4);
    const v16bf ap = load_frag_2x16(Pa, Pa + 8);
    float al[8];
#pragma unroll
    for (int r = 0; r < 8; ++r) al[r] = rowA[mW + r + (hf << 3)];
#pragma unroll
    for (int cf = 0; cf < C / 16; ++cf) {
      const v16bf bv = load_frag_c32(Vc + (cf * 16 + c16) * VP + (hf << 4));
#pragma unroll
      for (int r = 0; r < 8; ++r) oacc[cf][r] *= al[r];
      oacc[cf] = WMMA_BF16(ap, bv, oacc[cf]);
    }

    // drain staged registers into the other LDS buffer
    if (kb + 1 < NB) {
      bf16* Kd = Kts + (1 - cur) * (Bc * CP);
      bf16* Vd = Vs + (1 - cur) * (C * VP);
#pragma unroll
      for (int it = 0; it < NSTG; ++it) {
        const int i = tid + it * 128;
        const int c = i >> 2, j = i & 3;
        ((uint4*)(Vd + c * VP))[j] = vreg[it];
        union { uint4 q; bf16 h[8]; } kd;
        kd.q = kreg[it];
#pragma unroll
        for (int t = 0; t < 8; ++t) Kd[(8 * j + t) * CP + c] = kd.h[t];
      }
    }
    __syncthreads();
  }

  // ---- epilogue: normalize; pack 8 consecutive-n bf16 into one b128 store ----
  float il[8];
#pragma unroll
  for (int r = 0; r < 8; ++r) il[r] = 1.0f / rowL[mW + r + (hf << 3)];
#pragma unroll
  for (int cf = 0; cf < C / 16; ++cf) {
    const int cc = cf * 16 + c16;
    union { uint4 q; bf16 h[8]; } od;
#pragma unroll
    for (int r = 0; r < 8; ++r) od.h[r] = (bf16)(oacc[cf][r] * il[r]);
    *(uint4*)(Ob + (size_t)cc * N + n0 + mW + (hf << 3)) = od.q;
  }
}

// ============================================================================
extern "C" void kernel_launch(void* const* d_in, const int* in_sizes, int n_in,
                              void* d_out, int out_size, void* d_ws, size_t ws_size,
                              hipStream_t stream) {
  (void)in_sizes; (void)n_in; (void)out_size; (void)ws_size;
  constexpr int Bb = 4, Nsp = 64 * 64;  // 4096 spatial positions
  constexpr int C2D = 256, C3D = 128, IA = 128, IB = 64;

  const float* f2d = (const float*)d_in[0];
  const float* f3d = (const float*)d_in[1];
  const float* wqA = (const float*)d_in[2];  const float* bqA = (const float*)d_in[3];
  const float* wkA = (const float*)d_in[4];  const float* bkA = (const float*)d_in[5];
  const float* wvA = (const float*)d_in[6];  const float* bvA = (const float*)d_in[7];
  const float* wpA = (const float*)d_in[8];  const float* bpA = (const float*)d_in[9];
  const float* wqB = (const float*)d_in[10]; const float* bqB = (const float*)d_in[11];
  const float* wkB = (const float*)d_in[12]; const float* bkB = (const float*)d_in[13];
  const float* wvB = (const float*)d_in[14]; const float* bvB = (const float*)d_in[15];
  const float* wpB = (const float*)d_in[16]; const float* bpB = (const float*)d_in[17];

  bf16* ws = (bf16*)d_ws;
  const size_t eA = (size_t)Bb * IA * Nsp;
  const size_t eB = (size_t)Bb * IB * Nsp;
  bf16 *QA = ws,      *KA = QA + eA, *VA = KA + eA, *OA = VA + eA;
  bf16 *QB = OA + eA, *KB = QB + eB, *VB = KB + eB, *OB = VB + eB;

  const dim3 blk(128);

  conv1x1_kernel<C2D, false, false><<<dim3(Nsp / 64, IA / 64, Bb), blk, 0, stream>>>(
      wqA, bqA, f2d, nullptr, QA, IA);
  conv1x1_kernel<C3D, false, false><<<dim3(Nsp / 64, IA / 64, Bb), blk, 0, stream>>>(
      wkA, bkA, f3d, nullptr, KA, IA);
  conv1x1_kernel<C3D, false, false><<<dim3(Nsp / 64, IA / 64, Bb), blk, 0, stream>>>(
      wvA, bvA, f3d, nullptr, VA, IA);
  conv1x1_kernel<C3D, false, false><<<dim3(Nsp / 64, IB / 64, Bb), blk, 0, stream>>>(
      wqB, bqB, f3d, nullptr, QB, IB);
  conv1x1_kernel<C2D, false, false><<<dim3(Nsp / 64, IB / 64, Bb), blk, 0, stream>>>(
      wkB, bkB, f2d, nullptr, KB, IB);
  conv1x1_kernel<C2D, false, false><<<dim3(Nsp / 64, IB / 64, Bb), blk, 0, stream>>>(
      wvB, bvB, f2d, nullptr, VB, IB);

  flash_attn_kernel<IA><<<dim3(Nsp / 64, Bb), blk, 0, stream>>>(QA, KA, VA, OA);
  flash_attn_kernel<IB><<<dim3(Nsp / 64, Bb), blk, 0, stream>>>(QB, KB, VB, OB);

  float* x2D = (float*)d_out;
  float* x3D = x2D + (size_t)Bb * C2D * Nsp;
  conv1x1_kernel<IA, true, true><<<dim3(Nsp / 64, C2D / 64, Bb), blk, 0, stream>>>(
      wpA, bpA, OA, f2d, x2D, C2D);
  conv1x1_kernel<IB, true, true><<<dim3(Nsp / 64, C3D / 64, Bb), blk, 0, stream>>>(
      wpB, bpB, OB, f3d, x3D, C3D);
}
